// Net_7069516169728
// MI455X (gfx1250) — matrix operand order repaired and checked by reading, not compiled, not run on hardware
//
#include <hip/hip_runtime.h>

// ---------------------------------------------------------------------------
// GCN link-prediction forward (2x GCNConv + dot-product decode) for gfx1250.
// GEMMs use v_wmma_f32_16x16x32_bf16; aggregation is L2-resident scatter-add.
// ---------------------------------------------------------------------------

typedef __attribute__((ext_vector_type(16))) __bf16 v16bf;
typedef __attribute__((ext_vector_type(8)))  float  v8f;

struct BFrag32B { float4 a, b; };   // 32 bytes == v16bf

// ---------------------------- degree / norm --------------------------------

__global__ void init_deg_kernel(float* __restrict__ deg, int N) {
    int i = blockIdx.x * blockDim.x + threadIdx.x;
    if (i < N) deg[i] = 1.0f;                       // self-loop
}

__global__ void count_deg_kernel(const int* __restrict__ dst, float* __restrict__ deg, int E) {
    int e = blockIdx.x * blockDim.x + threadIdx.x;
    if (e < E) atomicAdd(&deg[dst[e]], 1.0f);
}

__global__ void rsqrt_kernel(const float* __restrict__ deg, float* __restrict__ dinv, int N) {
    int i = blockIdx.x * blockDim.x + threadIdx.x;
    if (i < N) dinv[i] = rsqrtf(deg[i]);            // deg >= 1 guaranteed
}

// ------------------------------ WMMA GEMM ----------------------------------
// Y[Nrows,NCOLS] = X[Nrows,128] @ W[128,NCOLS], fp32 in/out, bf16 compute,
// fp32 accumulate.  Block = 256 threads = 8 waves; each wave owns a 16-row
// stripe of a 128-row block and sweeps all 16-col tiles.  W is staged in LDS
// transposed (+8 element pad) so B-fragments are contiguous 32B ds loads.

template <int NCOLS>
__global__ __launch_bounds__(256) void gemm_wmma_kernel(
    const float* __restrict__ X, const float* __restrict__ W,
    float* __restrict__ Y, int Nrows) {
    constexpr int K   = 128;
    constexpr int LDW = K + 8;                       // bf16 elements, bank-conflict pad
    __shared__ __align__(32) __bf16 Wt[NCOLS * LDW];

    // Cooperative stage: W (row-major [K][NCOLS] f32) -> Wt[n][k] bf16
    for (int t = threadIdx.x; t < K * NCOLS; t += 256) {
        int k = t / NCOLS;
        int n = t - k * NCOLS;
        Wt[n * LDW + k] = (__bf16)W[t];
    }
    __syncthreads();

    const int wave = threadIdx.x >> 5;
    const int lane = threadIdx.x & 31;
    const int nn   = lane & 15;                      // column within tile / row M within A
    const int hi   = lane >> 4;                      // half-wave select
    const int m0   = blockIdx.x * 128 + wave * 16;
    if (m0 >= Nrows) return;                         // wave-uniform, EXEC stays full

    // A-fragments for this wave's 16 rows, all of K, loaded once.
    // Layout (16-bit A 16x32): lanes 0-15 hold K = kk .. kk+15, lanes 16-31
    // hold K = kk+16 .. kk+31 (contiguous, matching B).
    const float* xrow = X + (size_t)(m0 + nn) * K;
    v16bf afrag[4];
#pragma unroll
    for (int kb = 0; kb < 4; ++kb) {
        const float* p0 = xrow + kb * 32 + hi * 16;
        const float* p1 = p0 + 8;
        float4 f0 = *(const float4*)(p0);
        float4 f1 = *(const float4*)(p0 + 4);
        float4 f2 = *(const float4*)(p1);
        float4 f3 = *(const float4*)(p1 + 4);
        v16bf a;
        a[0]  = (__bf16)f0.x; a[1]  = (__bf16)f0.y; a[2]  = (__bf16)f0.z; a[3]  = (__bf16)f0.w;
        a[4]  = (__bf16)f1.x; a[5]  = (__bf16)f1.y; a[6]  = (__bf16)f1.z; a[7]  = (__bf16)f1.w;
        a[8]  = (__bf16)f2.x; a[9]  = (__bf16)f2.y; a[10] = (__bf16)f2.z; a[11] = (__bf16)f2.w;
        a[12] = (__bf16)f3.x; a[13] = (__bf16)f3.y; a[14] = (__bf16)f3.z; a[15] = (__bf16)f3.w;
        afrag[kb] = a;
    }

#pragma unroll
    for (int nt = 0; nt < NCOLS / 16; ++nt) {
        const int n0 = nt * 16;
        v8f acc = {};
#pragma unroll
        for (int kb = 0; kb < 4; ++kb) {
            // B frag in this code's contiguous k order (same as A): lanes 0-15
            // K=0..15, lanes 16-31 K=16..31 -> 32 contiguous bytes per lane.
            const __bf16* q = &Wt[(n0 + nn) * LDW + kb * 32 + hi * 16];
            BFrag32B raw = *(const BFrag32B*)q;      // 2x ds_load_b128 (16B aligned)
            v16bf bfr = __builtin_bit_cast(v16bf, raw);
            acc = __builtin_amdgcn_wmma_f32_16x16x32_bf16(
                false, afrag[kb], false, bfr, (short)0, acc, false, false);
        }
        // C/D layout: VGPR r -> M = r + 8*hi, N = lane&15
        float* yb = Y + (size_t)(m0 + hi * 8) * NCOLS + n0 + nn;
#pragma unroll
        for (int r = 0; r < 8; ++r) yb[(size_t)r * NCOLS] = acc[r];
    }
}

// -------------------- aggregation: self-loop init + scatter ----------------

template <int F>
__global__ void scale_self_kernel(const float* __restrict__ h, const float* __restrict__ dinv,
                                  float* __restrict__ out, int N) {
    long long idx   = (long long)blockIdx.x * blockDim.x + threadIdx.x;   // float4 units
    long long total = (long long)N * (F / 4);
    if (idx >= total) return;
    int   i = (int)(idx / (F / 4));
    float di = dinv[i];
    float s  = di * di;
    float4 v = ((const float4*)h)[idx];
    v.x *= s; v.y *= s; v.z *= s; v.w *= s;
    ((float4*)out)[idx] = v;
}

template <int F, int VEC>
__global__ void scatter_kernel(const float* __restrict__ h, const int* __restrict__ src,
                               const int* __restrict__ dst, const float* __restrict__ dinv,
                               float* __restrict__ out, int E) {
    int wid  = (int)(((long long)blockIdx.x * blockDim.x + threadIdx.x) >> 5);
    int lane = threadIdx.x & 31;
    if (wid >= E) return;
    int s = src[wid];
    int d = dst[wid];
    int eNext = wid + 512;                            // warm L2 ahead of the gather
    if (eNext < E) {
        __builtin_prefetch(h + (size_t)src[eNext] * F + lane * VEC, 0, 0);
    }
    float nrm = dinv[s] * dinv[d];
    const float* hp = h   + (size_t)s * F + lane * VEC;
    float*       op = out + (size_t)d * F + lane * VEC;
    if (VEC == 4) {
        float4 v = *(const float4*)hp;
        atomicAdd(op + 0, v.x * nrm);
        atomicAdd(op + 1, v.y * nrm);
        atomicAdd(op + 2, v.z * nrm);
        atomicAdd(op + 3, v.w * nrm);
    } else {
        float2 v = *(const float2*)hp;
        atomicAdd(op + 0, v.x * nrm);
        atomicAdd(op + 1, v.y * nrm);
    }
}

template <int F, bool RELU>
__global__ void bias_act_kernel(const float* __restrict__ in, const float* __restrict__ bias,
                                float* __restrict__ out, long long n) {
    long long i = (long long)blockIdx.x * blockDim.x + threadIdx.x;
    if (i >= n) return;
    float v = in[i] + bias[(int)(i & (F - 1))];
    out[i] = RELU ? fmaxf(v, 0.0f) : v;
}

// ------------------------------- decode ------------------------------------
// One wave per edge pair: 64-dim dot product, wave32 xor-shuffle reduction.

__global__ void decode_kernel(const float* __restrict__ z, const int* __restrict__ pos,
                              const int* __restrict__ neg, float* __restrict__ out, int Ep) {
    int wid  = (int)(((long long)blockIdx.x * blockDim.x + threadIdx.x) >> 5);
    int lane = threadIdx.x & 31;
    if (wid >= 2 * Ep) return;
    int a, b;
    if (wid < Ep) { a = pos[wid];      b = pos[Ep + wid]; }
    else          { a = neg[wid - Ep]; b = neg[wid];      }   // neg[Ep + (wid-Ep)]
    const float* za = z + (size_t)a * 64;
    const float* zb = z + (size_t)b * 64;
    float p = za[lane] * zb[lane] + za[lane + 32] * zb[lane + 32];
#pragma unroll
    for (int off = 16; off > 0; off >>= 1) p += __shfl_xor(p, off, 32);
    if (lane == 0) out[wid] = p;
}

// ------------------------------ launcher -----------------------------------

extern "C" void kernel_launch(void* const* d_in, const int* in_sizes, int n_in,
                              void* d_out, int out_size, void* d_ws, size_t ws_size,
                              hipStream_t stream) {
    const float* x   = (const float*)d_in[0];
    const int*   tpe = (const int*)d_in[1];
    const int*   pos = (const int*)d_in[2];
    const int*   neg = (const int*)d_in[3];
    const float* W1  = (const float*)d_in[4];
    const float* b1  = (const float*)d_in[5];
    const float* W2  = (const float*)d_in[6];
    const float* b2  = (const float*)d_in[7];

    const int N  = in_sizes[0] / 128;   // nodes (F=128)
    const int E  = in_sizes[1] / 2;     // train edges
    const int Ep = in_sizes[2] / 2;     // pos (=neg) decode edges
    const int* src = tpe;
    const int* dst = tpe + E;
    float* out = (float*)d_out;
    (void)n_in; (void)out_size; (void)ws_size;

    // Carve workspace (256B-aligned offsets).
    char* wsp = (char*)d_ws;
    auto carve = [&](size_t bytes) -> float* {
        float* p = (float*)wsp;
        wsp += (bytes + 255) & ~(size_t)255;
        return p;
    };
    float* deg  = carve((size_t)N * 4);
    float* dinv = carve((size_t)N * 4);
    float* bufH = carve((size_t)N * 128 * 4);  // hx -> h1 -> z
    float* bufA = carve((size_t)N * 128 * 4);  // agg1 -> hz
    float* bufC = carve((size_t)N * 64 * 4);   // agg2

    const int T = 256;
    auto cdiv = [](long long a, long long b) { return (int)((a + b - 1) / b); };

    // Degree + symmetric normalization
    init_deg_kernel<<<cdiv(N, T), T, 0, stream>>>(deg, N);
    count_deg_kernel<<<cdiv(E, T), T, 0, stream>>>(dst, deg, E);
    rsqrt_kernel<<<cdiv(N, T), T, 0, stream>>>(deg, dinv, N);

    // ---- Layer 1: h1 = relu(Agg(x @ W1) + b1) ----
    gemm_wmma_kernel<128><<<cdiv(N, 128), T, 0, stream>>>(x, W1, bufH, N);
    scale_self_kernel<128><<<cdiv((long long)N * 32, T), T, 0, stream>>>(bufH, dinv, bufA, N);
    scatter_kernel<128, 4><<<cdiv((long long)E, 8), T, 0, stream>>>(bufH, src, dst, dinv, bufA, E);
    bias_act_kernel<128, true><<<cdiv((long long)N * 128, T), T, 0, stream>>>(
        bufA, b1, bufH, (long long)N * 128);

    // ---- Layer 2: z = Agg(h1 @ W2) + b2 ----
    gemm_wmma_kernel<64><<<cdiv(N, 128), T, 0, stream>>>(bufH, W2, bufA, N);
    scale_self_kernel<64><<<cdiv((long long)N * 16, T), T, 0, stream>>>(bufA, dinv, bufC, N);
    scatter_kernel<64, 2><<<cdiv((long long)E, 8), T, 0, stream>>>(bufA, src, dst, dinv, bufC, E);
    bias_act_kernel<64, false><<<cdiv((long long)N * 64, T), T, 0, stream>>>(
        bufC, b2, bufH, (long long)N * 64);

    // ---- Decode: logits = sum(z[a] * z[b]) ----
    decode_kernel<<<cdiv((long long)2 * Ep, 8), T, 0, stream>>>(bufH, pos, neg, out, Ep);
}
